// RWKVMambaHybrid_22686017257877
// MI455X (gfx1250) — matrix operand order repaired and checked by reading, not compiled, not run on hardware
//
#include <hip/hip_runtime.h>
#include <hip/hip_bf16.h>

// ---------------------------------------------------------------------------
// RWKV/Mamba hybrid scan for gfx1250 (MI455X), chunk-parallel formulation:
//   GEMM1:  U = X @ Bmat^T                  (bf16 WMMA, 64x16/wave, U stored bf16)
//   GEMM2:  S_intra = U_blk @ W             (bf16 WMMA, 64x64/wave, TDM double-
//           W = block-Toeplitz of (A^T)^k    buffered LDS staging of A-panel,
//                                            W pre-swizzled to B-fragment order)
//   scan :  128-step cross-chunk carry on 8x64 state (1 workgroup)
//   GEMM3:  S += s_prev @ Pc                (bf16 WMMA)
//   GEMM4:  Y = S @ Cmat^T                  (bf16 WMMA, 16x64/wave)
//   h-mix:  3-phase chunked decay scan, y += h*D
// x (128MB) is L2-resident (192MB L2); HBM floor ~11-20us; scan latency
// collapsed from 4096 dependent steps to 128 + 64 tiny steps.
// ---------------------------------------------------------------------------

typedef __attribute__((ext_vector_type(16))) __bf16 v16bf;
typedef __attribute__((ext_vector_type(8)))  __bf16 v8bf;
typedef __attribute__((ext_vector_type(8)))  float  v8f;
typedef __attribute__((ext_vector_type(4)))  unsigned int u32x4;
typedef __attribute__((ext_vector_type(8)))  int i32x8;
typedef __attribute__((ext_vector_type(4)))  int i32x4;

#define WMMA_BF16(a, b, c) \
  __builtin_amdgcn_wmma_f32_16x16x32_bf16(false, (a), false, (b), (short)0, (c), false, false)

#if defined(__has_builtin)
#  if __has_builtin(__builtin_amdgcn_tensor_load_to_lds) && \
      __has_builtin(__builtin_amdgcn_s_wait_tensorcnt)
#    define USE_TDM 1
#  endif
#endif
#ifndef USE_TDM
#  define USE_TDM 0
#endif

constexpr int Bsz = 8, T = 4096, C = 1024, N = 64;
constexpr int LCH = 32;          // SSM chunk length
constexpr int NCH = T / LCH;     // 128 chunks
constexpr int KW  = LCH * N;     // 2048 lifted dim
constexpr int LH  = 64;          // token-mix chunk length
constexpr int NHC = T / LH;      // 64 chunks

// ---------------- WMMA fragment helpers (C/D per CDNA5 ISA 7.12.2; A/B in contiguous k = half*16+s) ------

// A 16x32 bf16 from fp32 row-major src: lane M=lane&15, half=lane>>4,
// VGPR r: K = half*16 + 2r  -> one 16-elem contiguous run.
static __device__ __forceinline__ v16bf load_a16x32_f32(const float* __restrict__ src,
                                                        int ld, int m0, int k0) {
  int lane = threadIdx.x & 31;
  const float* row = src + (size_t)(m0 + (lane & 15)) * ld;
  int half = (lane >> 4);
  v16bf a;
#pragma unroll
  for (int r = 0; r < 8; ++r) {
    int k = k0 + half * 16 + 2 * r;
    a[2 * r]     = (__bf16)row[k];
    a[2 * r + 1] = (__bf16)row[k + 1];
  }
  return a;
}

// B 32x16 bf16 from fp32 K-major src: element (k,n) = src[k*ld + n];
// lane N=lane&15, half=lane>>4, VGPR r: K = half*16 + 2r.
static __device__ __forceinline__ v16bf load_b32x16_kn_f32(const float* __restrict__ src,
                                                           int ld, int k0, int n0) {
  int lane = threadIdx.x & 31;
  int n = n0 + (lane & 15);
  int half = (lane >> 4);
  v16bf b;
#pragma unroll
  for (int r = 0; r < 8; ++r) {
    int k = k0 + half * 16 + 2 * r;
    b[2 * r]     = (__bf16)src[(size_t)k * ld + n];
    b[2 * r + 1] = (__bf16)src[(size_t)(k + 1) * ld + n];
  }
  return b;
}

// Same but src stored N-major (element (k,n) = src[n*ld+k]) -> contiguous in k.
static __device__ __forceinline__ v16bf load_b32x16_nk_f32(const float* __restrict__ src,
                                                           int ld, int k0, int n0) {
  int lane = threadIdx.x & 31;
  const float* row = src + (size_t)(n0 + (lane & 15)) * ld;
  int half = (lane >> 4);
  v16bf b;
#pragma unroll
  for (int r = 0; r < 8; ++r) {
    int k = k0 + half * 16 + 2 * r;
    b[2 * r]     = (__bf16)row[k];
    b[2 * r + 1] = (__bf16)row[k + 1];
  }
  return b;
}

// C/D f32 16x16: VGPR r -> row (r + 8*(lane>>4)), col = lane&15.
static __device__ __forceinline__ void store_c16x16(float* __restrict__ dst, int ld,
                                                    int m0, int n0, v8f acc) {
  int lane = threadIdx.x & 31;
  int n = n0 + (lane & 15);
  int mb = m0 + (lane >> 4) * 8;
#pragma unroll
  for (int r = 0; r < 8; ++r) dst[(size_t)(mb + r) * ld + n] = acc[r];
}

static __device__ __forceinline__ void store_c16x16_bf16(__bf16* __restrict__ dst, int ld,
                                                         int m0, int n0, v8f acc) {
  int lane = threadIdx.x & 31;
  int n = n0 + (lane & 15);
  int mb = m0 + (lane >> 4) * 8;
#pragma unroll
  for (int r = 0; r < 8; ++r) dst[(size_t)(mb + r) * ld + n] = (__bf16)acc[r];
}

static __device__ __forceinline__ v8f load_c16x16(const float* __restrict__ src, int ld,
                                                  int m0, int n0) {
  int lane = threadIdx.x & 31;
  int n = n0 + (lane & 15);
  int mb = m0 + (lane >> 4) * 8;
  v8f acc;
#pragma unroll
  for (int r = 0; r < 8; ++r) acc[r] = src[(size_t)(mb + r) * ld + n];
  return acc;
}

// ---------------- TDM issue: 128x32 bf16 A-panel -> LDS ---------------------

#if USE_TDM
static __device__ __forceinline__ void tdm_issue(const __bf16* base, int row0, int kt,
                                                 unsigned lds_addr) {
  // D# group0: count=1(load), lds_addr, 57-bit global byte addr, type=2
  unsigned long long ga =
      (unsigned long long)(size_t)(base + (size_t)row0 * KW + kt * 32);
  u32x4 g0;
  g0[0] = 1u;
  g0[1] = lds_addr;
  g0[2] = (unsigned)(ga & 0xffffffffu);
  g0[3] = (unsigned)((ga >> 32) & 0x1ffffffu) | 0x80000000u;
  // D# group1: data_size=2B; tensor_dim0=32, tensor_dim1=128; tile 32x128;
  // tensor_dim0_stride = KW elements (row stride of U). Rows pack densely in
  // LDS: row r at lds_addr + r*64 bytes.
  i32x8 g1;
  g1[0] = 0x00010000;        // data_size code 1 (2 bytes)
  g1[1] = (32 << 16);        // tensor_dim0[15:0] in bits 31:16
  g1[2] = (128 << 16);       // tensor_dim1[15:0]
  g1[3] = (32 << 16);        // tile_dim0
  g1[4] = 128;               // tile_dim1 (tile_dim2 = 0)
  g1[5] = KW;                // tensor_dim0_stride low 32
  g1[6] = 0;
  g1[7] = 0;
  i32x4 z4 = {0, 0, 0, 0};
#if __clang_major__ >= 23
  i32x8 z8 = {0, 0, 0, 0, 0, 0, 0, 0};
  __builtin_amdgcn_tensor_load_to_lds(g0, g1, z4, z4, z8, 0);
#else
  __builtin_amdgcn_tensor_load_to_lds(g0, g1, z4, z4, 0);
#endif
}
#endif

// ---------------- GEMM kernels ----------------------------------------------

// GEMM1: U[32768,64](bf16) = X[32768,1024] @ Bmat^T ; 64x16 per wave.
__global__ __launch_bounds__(256) void gemm1_u(const float* __restrict__ X,
                                               const float* __restrict__ Bm,
                                               __bf16* __restrict__ U) {
  int wave = blockIdx.x * 8 + (threadIdx.x >> 5);   // 2048 waves
  int mt = wave >> 2, nt = wave & 3;                // 512 x 4
  int m0 = mt * 64, n0 = nt * 16;
  v8f zero = {};
  v8f acc[4] = {zero, zero, zero, zero};
  for (int k0 = 0; k0 < 1024; k0 += 32) {
    v16bf b = load_b32x16_nk_f32(Bm, 1024, k0, n0);
#pragma unroll
    for (int i = 0; i < 4; ++i) {
      v16bf a = load_a16x32_f32(X, 1024, m0 + i * 16, k0);
      acc[i] = WMMA_BF16(a, b, acc[i]);
    }
  }
#pragma unroll
  for (int i = 0; i < 4; ++i) store_c16x16_bf16(U, 64, m0 + i * 16, n0, acc[i]);
}

// GEMM2: S[1024,2048](f32) = Ublk[1024,2048](bf16) @ W[2048,2048](bf16 swizzled)
// Block 128x256 (8 waves, 2x4), 64x64 per wave. A-panel (128x32 bf16 = 8KB)
// staged in LDS via TDM (double buffered); W read as contiguous B-fragments.
__global__ __launch_bounds__(256) void gemm2_intra(const __bf16* __restrict__ U,
                                                   const __bf16* __restrict__ Wsw,
                                                   float* __restrict__ S) {
  __shared__ __attribute__((aligned(32))) __bf16 sU[2][128 * 32];
  int w = threadIdx.x >> 5;
  int lane = threadIdx.x & 31;
  int half = lane >> 4;
  int bm = blockIdx.x >> 3, bn = blockIdx.x & 7;    // 8 x 8 blocks
  int wm = w >> 2, wn = w & 3;
  int row0 = bm * 128;                              // block A-panel rows
  int n0 = bn * 256 + wn * 64;                      // wave col origin

  v8f zero = {};
  v8f acc[4][4];
#pragma unroll
  for (int i = 0; i < 4; ++i)
#pragma unroll
    for (int j = 0; j < 4; ++j) acc[i][j] = zero;

  const int NK = KW / 32;                           // 64 k-steps
#if USE_TDM
  unsigned lds0 = (unsigned)(size_t)(&sU[0][0]);
  unsigned lds1 = (unsigned)(size_t)(&sU[1][0]);
  if (w == 0) tdm_issue(U, row0, 0, lds0);
#endif
  for (int kt = 0; kt < NK; ++kt) {
#if USE_TDM
    if (w == 0) {
      if (kt + 1 < NK) {
        tdm_issue(U, row0, kt + 1, ((kt + 1) & 1) ? lds1 : lds0);
        __builtin_amdgcn_s_wait_tensorcnt(1);       // current panel complete
      } else {
        __builtin_amdgcn_s_wait_tensorcnt(0);
      }
    }
#else
    {   // cooperative fallback: 256 threads x 32B = 8KB panel
      int boff = threadIdx.x * 16;                  // bf16 elements
      int r = boff >> 5, kk = boff & 31;
      *(v16bf*)(&sU[kt & 1][boff]) =
          *(const v16bf*)(U + (size_t)(row0 + r) * KW + kt * 32 + kk);
    }
#endif
    __syncthreads();
    const __bf16* sbuf = &sU[kt & 1][0];
    // A fragments from LDS: two contiguous 16B runs per lane (ds_load_b128 x2)
    v16bf afr[4];
#pragma unroll
    for (int i = 0; i < 4; ++i) {
      int rl = wm * 64 + i * 16 + (lane & 15);
      v8bf lo = *(const v8bf*)(sbuf + rl * 32 + half * 16);
      v8bf hi = *(const v8bf*)(sbuf + rl * 32 + half * 16 + 8);
      v16bf a;
#pragma unroll
      for (int e = 0; e < 8; ++e) { a[e] = lo[e]; a[8 + e] = hi[e]; }
      afr[i] = a;
    }
    // B fragments: 32B contiguous per lane from pre-swizzled W
#pragma unroll
    for (int j = 0; j < 4; ++j) {
      int ntile = (n0 >> 4) + j;
      v16bf bfr = *(const v16bf*)(Wsw + ((size_t)(kt * 128 + ntile) * 32 + lane) * 16);
#pragma unroll
      for (int i = 0; i < 4; ++i) acc[i][j] = WMMA_BF16(afr[i], bfr, acc[i][j]);
    }
    __syncthreads();
  }
  int m0 = row0 + wm * 64;
#pragma unroll
  for (int i = 0; i < 4; ++i)
#pragma unroll
    for (int j = 0; j < 4; ++j)
      store_c16x16(S, KW, m0 + i * 16, n0 + j * 16, acc[i][j]);
}

// GEMM3: S[1024,2048] += Sprev[1024,64] @ Pc[64,2048] ; 16x16 per wave.
__global__ __launch_bounds__(256) void gemm3_carry(const float* __restrict__ Sprev,
                                                   const float* __restrict__ Pc,
                                                   float* __restrict__ S) {
  int wave = blockIdx.x * 8 + (threadIdx.x >> 5);
  int mt = wave >> 7, nt = wave & 127;
  int m0 = mt * 16, n0 = nt * 16;
  v8f acc = load_c16x16(S, KW, m0, n0);
  for (int k0 = 0; k0 < 64; k0 += 32) {
    v16bf a = load_a16x32_f32(Sprev, 64, m0, k0);
    v16bf b = load_b32x16_kn_f32(Pc, KW, k0, n0);
    acc = WMMA_BF16(a, b, acc);
  }
  store_c16x16(S, KW, m0, n0, acc);
}

// GEMM4: Y[32768,1024] = S[32768,64] @ Cmat^T ; 16x64 per wave.
__global__ __launch_bounds__(256) void gemm4_y(const float* __restrict__ S,
                                               const float* __restrict__ Cm,
                                               float* __restrict__ Y) {
  int wave = blockIdx.x * 8 + (threadIdx.x >> 5);   // 32768 waves
  int mt = wave >> 4, nt = wave & 15;               // 2048 x 16
  int m0 = mt * 16, n0 = nt * 64;
  v8f zero = {};
  v8f acc[4] = {zero, zero, zero, zero};
  for (int k0 = 0; k0 < 64; k0 += 32) {
    v16bf a = load_a16x32_f32(S, 64, m0, k0);
#pragma unroll
    for (int j = 0; j < 4; ++j) {
      v16bf b = load_b32x16_nk_f32(Cm, 64, k0, n0 + j * 16);
      acc[j] = WMMA_BF16(a, b, acc[j]);
    }
  }
#pragma unroll
  for (int j = 0; j < 4; ++j) store_c16x16(Y, 1024, m0, n0 + j * 16, acc[j]);
}

// ---------------- P / W / Pc construction ----------------------------------

// P[k] = (A^T)^k, k=0..LCH (fp32 64x64 each).
__global__ __launch_bounds__(1024) void buildP_kernel(const float* __restrict__ A,
                                                      float* __restrict__ P) {
  __shared__ float sA[64 * 64];
  __shared__ float sP[64 * 64];
  int tid = threadIdx.x;
  for (int e = tid; e < 4096; e += 1024) {
    sA[e] = A[e];
    float v = ((e >> 6) == (e & 63)) ? 1.f : 0.f;
    sP[e] = v;
    P[e] = v;
  }
  __syncthreads();
  for (int k = 1; k <= LCH; ++k) {
    float vals[4];
#pragma unroll
    for (int j = 0; j < 4; ++j) {
      int e = tid + j * 1024;
      int m = e >> 6, n = e & 63;
      float a = 0.f;
#pragma unroll
      for (int l = 0; l < 64; ++l) a += sP[m * 64 + l] * sA[n * 64 + l];
      vals[j] = a;
    }
    __syncthreads();
#pragma unroll
    for (int j = 0; j < 4; ++j) {
      int e = tid + j * 1024;
      sP[e] = vals[j];
      P[k * 4096 + e] = vals[j];
    }
    __syncthreads();
  }
}

// W pre-swizzled to B-fragment order: Wsw[((kt*128+nt)*32+lane)*16+e] holds
// W[k, col] with k = kt*32 + (lane>>4)*16 + e, col = nt*16 + (lane&15);
// W[k=(j,m), col=(i,n)] = (i>=j) ? P[i-j][m,n] : 0.
__global__ __launch_bounds__(256) void buildW_kernel(const float* __restrict__ P,
                                                     __bf16* __restrict__ Wsw) {
  int idx = blockIdx.x * 256 + threadIdx.x;         // 4M
  int e = idx & 15;
  int lane = (idx >> 4) & 31;
  int nt = (idx >> 9) & 127;
  int kt = idx >> 16;
  int k = kt * 32 + (lane >> 4) * 16 + e;
  int col = nt * 16 + (lane & 15);
  int j = k >> 6, m = k & 63;
  int i = col >> 6, n = col & 63;
  float v = (i >= j) ? P[(i - j) * 4096 + m * 64 + n] : 0.f;
  Wsw[idx] = (__bf16)v;
}

// Pc[m,(i,n)] = P[i+1][m,n]   (fp32 [64,2048])
__global__ __launch_bounds__(256) void buildPc_kernel(const float* __restrict__ P,
                                                      float* __restrict__ Pc) {
  int idx = blockIdx.x * 256 + threadIdx.x;         // 131072
  int col = idx & (KW - 1), m = idx >> 11;
  int i = col >> 6, n = col & 63;
  Pc[idx] = P[(i + 1) * 4096 + m * 64 + n];
}

// Cross-chunk carry: s_prev(c+1) = s_prev(c) @ P_L + S_intra[chunk end].
__global__ __launch_bounds__(512) void carry_scan_kernel(const float* __restrict__ P,
                                                         const float* __restrict__ Sintra,
                                                         float* __restrict__ Sprev) {
  __shared__ float s[Bsz * 64];
  int tid = threadIdx.x;
  int b = tid >> 6, n = tid & 63;
  const float* PL = P + LCH * 4096;
  float plcol[64];
#pragma unroll
  for (int m = 0; m < 64; ++m) plcol[m] = PL[m * 64 + n];
  s[tid] = 0.f;
  __syncthreads();
  for (int c = 0; c < NCH; ++c) {
    Sprev[(b * NCH + c) * 64 + n] = s[tid];
    float a = Sintra[((size_t)(b * T) + c * LCH + (LCH - 1)) * 64 + n];
#pragma unroll
    for (int m = 0; m < 64; ++m) a += s[b * 64 + m] * plcol[m];
    __syncthreads();
    s[tid] = a;
    __syncthreads();
  }
}

// ---------------- token-mix decay scan (3-phase) ----------------------------

__global__ __launch_bounds__(256) void hscanA_kernel(const float* __restrict__ x,
                                                     const float* __restrict__ wmix,
                                                     float* __restrict__ E) {
  int idx = blockIdx.x * 256 + threadIdx.x;         // 524288
  int c = idx & (C - 1);
  int ch = (idx >> 10) & (NHC - 1);
  int b = idx >> 16;
  float w = wmix[c], iw = 1.f - w;
  const float* xp = x + ((size_t)(b * T + ch * LH)) * C + c;
  float h = 0.f;
  for (int i = 0; i < LH; ++i) h = w * h + iw * xp[(size_t)i * C];
  E[(b * NHC + ch) * C + c] = h;
}

__global__ __launch_bounds__(256) void hscanB_kernel(const float* __restrict__ E,
                                                     const float* __restrict__ wmix,
                                                     float* __restrict__ Hin) {
  int idx = blockIdx.x * 256 + threadIdx.x;         // 8192
  int c = idx & (C - 1), b = idx >> 10;
  float w = wmix[c];
  float wp = w;
#pragma unroll
  for (int q = 0; q < 6; ++q) wp *= wp;             // w^64
  float h = 0.f;
  for (int ch = 0; ch < NHC; ++ch) {
    Hin[(b * NHC + ch) * C + c] = h;
    h = wp * h + E[(b * NHC + ch) * C + c];
  }
}

__global__ __launch_bounds__(256) void haddY_kernel(const float* __restrict__ x,
                                                    const float* __restrict__ wmix,
                                                    const float* __restrict__ Dv,
                                                    const float* __restrict__ Hin,
                                                    float* __restrict__ y) {
  int idx = blockIdx.x * 256 + threadIdx.x;         // 524288
  int c = idx & (C - 1);
  int ch = (idx >> 10) & (NHC - 1);
  int b = idx >> 16;
  float w = wmix[c], iw = 1.f - w, d = Dv[c];
  float h = Hin[(b * NHC + ch) * C + c];
  const float* xp = x + ((size_t)(b * T + ch * LH)) * C + c;
  float* yp = y + ((size_t)(b * T + ch * LH)) * C + c;
  for (int i = 0; i < LH; ++i) {
    h = w * h + iw * xp[(size_t)i * C];
    yp[(size_t)i * C] += h * d;
  }
}

// ---------------- host launcher --------------------------------------------

extern "C" void kernel_launch(void* const* d_in, const int* in_sizes, int n_in,
                              void* d_out, int out_size, void* d_ws, size_t ws_size,
                              hipStream_t stream) {
  const float* x    = (const float*)d_in[0];   // [8,4096,1024]
  const float* wmix = (const float*)d_in[1];   // [1024]
  const float* A    = (const float*)d_in[2];   // [64,64]
  const float* Bm   = (const float*)d_in[3];   // [64,1024]
  const float* Cm   = (const float*)d_in[4];   // [1024,64]
  const float* Dv   = (const float*)d_in[5];   // [1024]
  float* y = (float*)d_out;                    // [8,4096,1024]

  // workspace layout (~30 MB), fp32 region first (keeps bf16 region 32B aligned)
  float* Sbuf  = (float*)d_ws;                          // 32768*64
  float* P     = Sbuf + (size_t)Bsz * T * N;            // 33*4096
  float* Pc    = P + (size_t)(LCH + 1) * 4096;          // 64*2048
  float* Sprev = Pc + (size_t)64 * KW;                  // 8*128*64
  float* E     = Sprev + (size_t)Bsz * NCH * N;         // 8*64*1024
  float* Hin   = E + (size_t)Bsz * NHC * C;             // 8*64*1024
  __bf16* Ubf  = (__bf16*)(Hin + (size_t)Bsz * NHC * C);// 32768*64 bf16
  __bf16* Wsw  = Ubf + (size_t)Bsz * T * N;             // 2048*2048 bf16

  // SSM pipeline (stream order provides all dependencies)
  gemm1_u<<<256, 256, 0, stream>>>(x, Bm, Ubf);                // U = X @ B^T
  buildP_kernel<<<1, 1024, 0, stream>>>(A, P);                 // P_k = (A^T)^k
  buildW_kernel<<<(KW * KW) / 256, 256, 0, stream>>>(P, Wsw);  // swizzled Toeplitz
  buildPc_kernel<<<(64 * KW) / 256, 256, 0, stream>>>(P, Pc);  // carry projector
  gemm2_intra<<<64, 256, 0, stream>>>(Ubf, Wsw, Sbuf);         // intra-chunk conv
  carry_scan_kernel<<<1, 512, 0, stream>>>(P, Sbuf, Sprev);    // 128-step scan
  gemm3_carry<<<1024, 256, 0, stream>>>(Sprev, Pc, Sbuf);      // S += carry
  gemm4_y<<<4096, 256, 0, stream>>>(Sbuf, Cm, y);              // Y = S @ C^T

  // token-mix pipeline, fused add into y
  hscanA_kernel<<<2048, 256, 0, stream>>>(x, wmix, E);
  hscanB_kernel<<<32, 256, 0, stream>>>(E, wmix, Hin);
  haddY_kernel<<<2048, 256, 0, stream>>>(x, wmix, Dv, Hin, y);
}
